// SFNAttention_53815940219184
// MI455X (gfx1250) — matrix operand order repaired and checked by reading, not compiled, not run on hardware
//
#include <hip/hip_runtime.h>
#include <hip/hip_bf16.h>

// ---------------- problem constants ----------------
#define D_MODEL 1024
#define NHEAD   16
#define HEAD_DIM 64
#define BATCH   2
#define SEQ     2048
#define MROWS   (BATCH * SEQ)          // 4096
#define SCALE_ATT 0.125f               // 1/sqrt(64)

typedef _Float16 v8h  __attribute__((ext_vector_type(8)));
typedef _Float16 v16h __attribute__((ext_vector_type(16)));
typedef float    v8f  __attribute__((ext_vector_type(8)));

// ---------------- CDNA5 async global->LDS copy (16 bytes per call) ----------------
#if defined(__has_builtin)
#if __has_builtin(__builtin_amdgcn_global_load_async_to_lds_b128)
#define HAVE_ASYNC_LDS 1
#endif
#if __has_builtin(__builtin_amdgcn_s_wait_asynccnt)
#define HAVE_WAIT_ASYNC 1
#endif
#endif

typedef int v4i_ __attribute__((vector_size(16)));
typedef __attribute__((address_space(1))) v4i_ g_v4i;   // global 128-bit payload
typedef __attribute__((address_space(3))) v4i_ l_v4i;   // LDS    128-bit payload

__device__ __forceinline__ void async_cp16(const _Float16* g, _Float16* l) {
#if defined(HAVE_ASYNC_LDS)
    __builtin_amdgcn_global_load_async_to_lds_b128((g_v4i*)g, (l_v4i*)l, 0, 0);
#else
    *(v8h*)l = *(const v8h*)g;         // synchronous fallback
#endif
}

__device__ __forceinline__ void wait_async() {
#if defined(HAVE_ASYNC_LDS)
#if defined(HAVE_WAIT_ASYNC)
    __builtin_amdgcn_s_wait_asynccnt(0);
#else
    asm volatile("s_wait_asynccnt 0" ::: "memory");
#endif
#endif
}

// Concatenate two 16-byte half-chunks into one 32-byte A/B fragment.
__device__ __forceinline__ v16h cat8(v8h lo, v8h hi) {
    return __builtin_shufflevector(lo, hi, 0,1,2,3,4,5,6,7,8,9,10,11,12,13,14,15);
}

// A-fragment (16x32 f16, MxK): lane = (row | hi<<4); caller passes p = rowBase + hi*16.
__device__ __forceinline__ v16h load_afrag(const _Float16* p) {
    v8h lo = *(const v8h*)(p);
    v8h hi = *(const v8h*)(p + 8);
    return cat8(lo, hi);
}

__device__ __forceinline__ v8f wmma_f16(v16h a, v16h b, v8f c) {
    return __builtin_amdgcn_wmma_f32_16x16x32_f16(false, a, false, b, (short)0, c, false, false);
}

// ---------------- kernel 1: f32 -> f16 convert ----------------
__global__ void cvt_f2h(const float* __restrict__ in, _Float16* __restrict__ out, int n) {
    int i = blockIdx.x * blockDim.x + threadIdx.x;
    int stride = gridDim.x * blockDim.x;
    for (; i < n; i += stride) out[i] = (_Float16)in[i];
}

// ---------------- kernel 2: C[M,N] = A[M,K] @ W[N,K]^T + bias ----------------
// Block = 8 waves, all sharing one 16-row A panel (staged in LDS via async copy).
// Each wave owns a 16x64 tile: 1 LDS A-frag reused across 4 global W B-frags per K step.
// Grid must be exactly (M/16)*(N/64)/8 blocks of 256 threads.
__global__ void gemm_wmma(const _Float16* __restrict__ A, const _Float16* __restrict__ W,
                          const float* __restrict__ bias, float* __restrict__ C,
                          int M, int N, int K) {
    __shared__ _Float16 aTile[16 * 1024];            // 32 KB: one 16 x K panel (K<=1024)
    const int w    = threadIdx.x >> 5;
    const int lane = threadIdx.x & 31;
    const int col  = lane & 15;
    const int hif  = lane >> 4;
    const int ntiles_n    = N >> 6;                  // 64-wide wave tiles
    const int blocks_perm = ntiles_n >> 3;           // blocks covering one m-tile
    const int mt = blockIdx.x / blocks_perm;
    const int nt = (blockIdx.x % blocks_perm) * 8 + w;
    const int m0 = mt << 4;
    const int n0 = nt << 6;

    // ---- stage contiguous 16xK A panel into LDS (async): 256 thr x 16B x (K/128) iters
    {
        const _Float16* apanel = A + (size_t)m0 * K; // rows m0..m0+15 are contiguous
        const int iters = (16 * K) / (256 * 8);
        for (int it = 0; it < iters; ++it) {
            const int flat = (it * 256 + threadIdx.x) * 8;
            async_cp16(apanel + flat, aTile + flat);
        }
        wait_async();
        __syncthreads();
    }

    v8f acc0 = {}, acc1 = {}, acc2 = {}, acc3 = {};
    const _Float16* arow = aTile + col * K;          // LDS row for this lane

    for (int k0 = 0; k0 < K; k0 += 32) {
        v16h af = load_afrag(arow + k0 + hif * 16);  // ds_load_b128 x2
        const _Float16* wbase = W + (size_t)(n0 + col) * K + k0 + hif * 16;
        __builtin_prefetch(wbase + 128, 0, 1);       // global_prefetch_b8 on W stream
        v16h b0 = *(const v16h*)(wbase);
        v16h b1 = *(const v16h*)(wbase + (size_t)16 * K);
        v16h b2 = *(const v16h*)(wbase + (size_t)32 * K);
        v16h b3 = *(const v16h*)(wbase + (size_t)48 * K);
        acc0 = wmma_f16(af, b0, acc0);
        acc1 = wmma_f16(af, b1, acc1);
        acc2 = wmma_f16(af, b2, acc2);
        acc3 = wmma_f16(af, b3, acc3);
    }
    const float bv0 = bias[n0 +  0 + col];
    const float bv1 = bias[n0 + 16 + col];
    const float bv2 = bias[n0 + 32 + col];
    const float bv3 = bias[n0 + 48 + col];
#pragma unroll
    for (int j = 0; j < 8; ++j) {
        const int row = m0 + j + hif * 8;            // C layout: VGPR j -> row j (lo) / j+8 (hi)
        float* crow = C + (size_t)row * N + n0;
        crow[ 0 + col] = acc0[j] + bv0;
        crow[16 + col] = acc1[j] + bv1;
        crow[32 + col] = acc2[j] + bv2;
        crow[48 + col] = acc3[j] + bv3;
    }
}

// ---------------- kernel 3: RMS-norm (+gain) + ScaleAndFire quantize ----------------
__device__ __forceinline__ float sfn(float y) {
    float q = rintf(y * 2.0f);                       // STEP = 0.5
    q = fminf(fmaxf(q, -8.0f), 8.0f);                // SF_LEVELS = 8
    return q * 0.5f;
}

// mode 0: q/k -> rmsnorm over head_dim, * gain, SFN, write [BH, L, 64] row-major.
// mode 1: v   -> SFN only, write TRANSPOSED [BH, 64, L].
__global__ void qkv_post(const float* __restrict__ proj, const float* __restrict__ gain,
                         _Float16* __restrict__ dst, int mode) {
    const int lane = threadIdx.x & 31;
    const int wid  = blockIdx.x * (blockDim.x >> 5) + (threadIdx.x >> 5);
    if (wid >= MROWS * NHEAD) return;
    const int m    = wid >> 4;
    const int h    = wid & 15;
    const int b    = m >> 11;
    const int lpos = m & (SEQ - 1);
    const int bh   = b * NHEAD + h;

    const float* src = proj + (size_t)m * D_MODEL + h * HEAD_DIM;
    float v0 = src[lane];
    float v1 = src[lane + 32];

    if (mode == 0) {
        float ss = v0 * v0 + v1 * v1;
        ss += __shfl_xor(ss, 1, 32);
        ss += __shfl_xor(ss, 2, 32);
        ss += __shfl_xor(ss, 4, 32);
        ss += __shfl_xor(ss, 8, 32);
        ss += __shfl_xor(ss, 16, 32);
        const float r = rsqrtf(ss * (1.0f / 64.0f) + 1e-6f);
        v0 = v0 * r * gain[lane];
        v1 = v1 * r * gain[lane + 32];
        _Float16* drow = dst + ((size_t)bh * SEQ + lpos) * HEAD_DIM;
        drow[lane]      = (_Float16)sfn(v0);
        drow[lane + 32] = (_Float16)sfn(v1);
    } else {
        dst[((size_t)bh * HEAD_DIM + lane)      * SEQ + lpos] = (_Float16)sfn(v0);
        dst[((size_t)bh * HEAD_DIM + lane + 32) * SEQ + lpos] = (_Float16)sfn(v1);
    }
}

// ---------------- kernel 4: flash attention ----------------
// Block = 8 waves sharing one (b,h); each wave owns one 16-query tile.
// Per 32-key block: async-stage K tile (32x64) and V^T tile (64x32) into LDS,
// then all 8 waves compute from LDS. Grid must be exactly B*H*(L/16)/8 = 512 blocks.
__global__ void attn_wmma(const _Float16* __restrict__ qh, const _Float16* __restrict__ kh,
                          const _Float16* __restrict__ vt, _Float16* __restrict__ oh) {
    __shared__ _Float16 kTile[32 * 64];              // 4 KB: keys kb..kb+31, d=0..63
    __shared__ _Float16 vTile[64 * 32];              // 4 KB: V^T rows d=0..63, keys kb..kb+31
    __shared__ _Float16 plds[8 * 16 * 32];           // 8 KB: per-wave P tiles
    const int w    = threadIdx.x >> 5;
    const int lane = threadIdx.x & 31;
    const int col  = lane & 15;
    const int hif  = lane >> 4;
    const int hi8  = hif * 8;
    const int wid  = blockIdx.x * 8 + w;
    const int qt = wid & (SEQ / 16 - 1);             // 128 query tiles per (b,h)
    const int bh = wid >> 7;                         // uniform across block (8 | 128)
    const int b  = bh >> 4;
    const int h  = bh & 15;

    _Float16* lds = plds + w * 512;

    // Q fragments for K-dim chunks 0..31 and 32..63 (registers for the whole sweep)
    const _Float16* qrow = qh + ((size_t)bh * SEQ + qt * 16 + col) * HEAD_DIM;
    const v16h qa0 = load_afrag(qrow +  0 + hif * 16);
    const v16h qa1 = load_afrag(qrow + 32 + hif * 16);

    v8f acc0 = {}, acc1 = {}, acc2 = {}, acc3 = {};
    float run_m[8], run_l[8];
#pragma unroll
    for (int j = 0; j < 8; ++j) { run_m[j] = -1e30f; run_l[j] = 0.0f; }

    const _Float16* kbase = kh + (size_t)bh * SEQ * HEAD_DIM;
    const _Float16* vbase = vt + (size_t)bh * HEAD_DIM * SEQ;

    for (int kb = 0; kb < SEQ; kb += 32) {
        // ---- cooperative async staging: K tile is one contiguous 4 KB block
        async_cp16(kbase + (size_t)kb * HEAD_DIM + threadIdx.x * 8, kTile + threadIdx.x * 8);
        // V^T tile: 64 rows x 32 keys (row stride SEQ) -> 4 chunks of 8 halves per row
        {
            const int r = threadIdx.x >> 2;          // d row 0..63
            const int c = (threadIdx.x & 3) * 8;     // key chunk
            async_cp16(vbase + (size_t)r * SEQ + kb + c, vTile + r * 32 + c);
        }
        wait_async();
        __syncthreads();

        // ---- scores S[16q x 32k] = Q @ K^T (B-frags from LDS)
        v8f s0 = {}, s1 = {};
        {
            const _Float16* k0p = kTile + ( 0 + col) * HEAD_DIM + hif * 16;
            const _Float16* k1p = kTile + (16 + col) * HEAD_DIM + hif * 16;
            v16h kb00 = *(const v16h*)(k0p);
            v16h kb01 = *(const v16h*)(k0p + 32);
            v16h kb10 = *(const v16h*)(k1p);
            v16h kb11 = *(const v16h*)(k1p + 32);
            s0 = wmma_f16(qa0, kb00, s0);
            s0 = wmma_f16(qa1, kb01, s0);
            s1 = wmma_f16(qa0, kb10, s1);
            s1 = wmma_f16(qa1, kb11, s1);
        }
        // ---- online softmax per row (VGPR j <-> row j + 8*hif)
#pragma unroll
        for (int j = 0; j < 8; ++j) {
            float sv0 = s0[j] * SCALE_ATT;
            float sv1 = s1[j] * SCALE_ATT;
            float mx = fmaxf(sv0, sv1);
            mx = fmaxf(mx, __shfl_xor(mx, 1, 16));
            mx = fmaxf(mx, __shfl_xor(mx, 2, 16));
            mx = fmaxf(mx, __shfl_xor(mx, 4, 16));
            mx = fmaxf(mx, __shfl_xor(mx, 8, 16));
            const float mnew  = fmaxf(run_m[j], mx);
            const float alpha = __expf(run_m[j] - mnew);
            run_m[j] = mnew;
            const float p0 = __expf(sv0 - mnew);
            const float p1 = __expf(sv1 - mnew);
            float ps = p0 + p1;
            ps += __shfl_xor(ps, 1, 16);
            ps += __shfl_xor(ps, 2, 16);
            ps += __shfl_xor(ps, 4, 16);
            ps += __shfl_xor(ps, 8, 16);
            run_l[j] = run_l[j] * alpha + ps;
            acc0[j] *= alpha; acc1[j] *= alpha; acc2[j] *= alpha; acc3[j] *= alpha;
            const int r = j + hi8;
            lds[r * 32 +      col] = (_Float16)p0;
            lds[r * 32 + 16 + col] = (_Float16)p1;
        }
        asm volatile("s_wait_dscnt 0" ::: "memory"); // same-wave LDS RAW fence

        // ---- reload P as A-fragment (16 q-rows x 32 keys)
        const v16h pa = load_afrag(lds + col * 32 + hif * 16);

        // ---- out += P @ V (B-frags from staged V^T tile)
        const _Float16* vrow = vTile + col * 32 + hif * 16;
        v16h vb0 = *(const v16h*)(vrow);
        v16h vb1 = *(const v16h*)(vrow + 16 * 32);
        v16h vb2 = *(const v16h*)(vrow + 32 * 32);
        v16h vb3 = *(const v16h*)(vrow + 48 * 32);
        acc0 = wmma_f16(pa, vb0, acc0);
        acc1 = wmma_f16(pa, vb1, acc1);
        acc2 = wmma_f16(pa, vb2, acc2);
        acc3 = wmma_f16(pa, vb3, acc3);

        __syncthreads();                             // protect tiles before next staging
    }

    // ---- normalize and write O as f16 [M, D] for the final projection GEMM
#pragma unroll
    for (int j = 0; j < 8; ++j) {
        const float inv = 1.0f / run_l[j];
        const int m = b * SEQ + qt * 16 + j + hi8;
        _Float16* orow = oh + (size_t)m * D_MODEL + h * HEAD_DIM;
        orow[ 0 + col] = (_Float16)(acc0[j] * inv);
        orow[16 + col] = (_Float16)(acc1[j] * inv);
        orow[32 + col] = (_Float16)(acc2[j] * inv);
        orow[48 + col] = (_Float16)(acc3[j] * inv);
    }
}

// ---------------- host-side orchestration ----------------
extern "C" void kernel_launch(void* const* d_in, const int* in_sizes, int n_in,
                              void* d_out, int out_size, void* d_ws, size_t ws_size,
                              hipStream_t stream) {
    const float* x  = (const float*)d_in[0];
    const float* wq = (const float*)d_in[1];
    const float* bq = (const float*)d_in[2];
    const float* wk = (const float*)d_in[3];
    const float* bk = (const float*)d_in[4];
    const float* wv = (const float*)d_in[5];
    const float* bv = (const float*)d_in[6];
    const float* wo = (const float*)d_in[7];
    const float* bo = (const float*)d_in[8];
    const float* gq = (const float*)d_in[9];
    const float* gk = (const float*)d_in[10];

    char* ws = (char*)d_ws;
    const size_t MB = 1024ull * 1024ull;
    _Float16* xh  = (_Float16*)(ws);             //  8 MB: x f16 [4096,1024]
    _Float16* wqh = (_Float16*)(ws +  8 * MB);   //  2 MB each
    _Float16* wkh = (_Float16*)(ws + 10 * MB);
    _Float16* wvh = (_Float16*)(ws + 12 * MB);
    _Float16* woh = (_Float16*)(ws + 14 * MB);
    float*    pq  = (float*)   (ws + 16 * MB);   // 16 MB raw q proj
    float*    pk  = (float*)   (ws + 32 * MB);   // 16 MB raw k proj
    float*    pv  = (float*)   (ws + 48 * MB);   // 16 MB raw v proj
    _Float16* qh  = (_Float16*)(ws + 64 * MB);   //  8 MB [BH,L,64]
    _Float16* kh  = (_Float16*)(ws + 72 * MB);   //  8 MB [BH,L,64]
    _Float16* vt  = (_Float16*)(ws + 80 * MB);   //  8 MB [BH,64,L]
    _Float16* oh  = (_Float16*)(ws + 16 * MB);   // alias pq (dead by then): 8 MB [M,D]

    // 1) f32 -> f16 conversions
    cvt_f2h<<<2048, 256, 0, stream>>>(x,  xh,  MROWS * D_MODEL);
    cvt_f2h<<<1024, 256, 0, stream>>>(wq, wqh, D_MODEL * D_MODEL);
    cvt_f2h<<<1024, 256, 0, stream>>>(wk, wkh, D_MODEL * D_MODEL);
    cvt_f2h<<<1024, 256, 0, stream>>>(wv, wvh, D_MODEL * D_MODEL);
    cvt_f2h<<<1024, 256, 0, stream>>>(wo, woh, D_MODEL * D_MODEL);

    // 2) projections: (4096 wave tiles / 8) = 512 blocks (exact, no guard)
    gemm_wmma<<<512, 256, 0, stream>>>(xh, wqh, bq, pq, MROWS, D_MODEL, D_MODEL);
    gemm_wmma<<<512, 256, 0, stream>>>(xh, wkh, bk, pk, MROWS, D_MODEL, D_MODEL);
    gemm_wmma<<<512, 256, 0, stream>>>(xh, wvh, bv, pv, MROWS, D_MODEL, D_MODEL);

    // 3) RMS-norm + SFN quantize
    qkv_post<<<8192, 256, 0, stream>>>(pq, gq, qh, 0);
    qkv_post<<<8192, 256, 0, stream>>>(pk, gk, kh, 0);
    qkv_post<<<8192, 256, 0, stream>>>(pv, gq, vt, 1);

    // 4) flash attention: 512 blocks (exact)
    attn_wmma<<<512, 256, 0, stream>>>(qh, kh, vt, oh);

    // 5) output projection -> f32 d_out
    gemm_wmma<<<512, 256, 0, stream>>>(oh, woh, bo, (float*)d_out, MROWS, D_MODEL, D_MODEL);
}
